// TrulyPureTransformer_62380105007450
// MI455X (gfx1250) — matrix operand order repaired and checked by reading, not compiled, not run on hardware
//
#include <hip/hip_runtime.h>

typedef __attribute__((ext_vector_type(16))) _Float16     v16h;
typedef __attribute__((ext_vector_type(8)))  _Float16     v8h;
typedef __attribute__((ext_vector_type(8)))  float        v8f;
typedef __attribute__((ext_vector_type(4)))  unsigned int u32x4;
typedef __attribute__((ext_vector_type(8)))  int          i32x8;
typedef __attribute__((ext_vector_type(4)))  int          i32x4;

#define DIMN      1024
#define HEADS     16
#define HEAD_DIM  64
#define LAYERS    8
#define HIDDEN    4096
#define VOCAB     266
#define BSZ       4
#define LSEQ      2048
#define TTOK      (BSZ * LSEQ)
#define ATT_SCALE 0.125f
#define LN_EPS    1e-5f

// ---------------------------------------------------------------------------
// TDM: issue a 2D f16 tile load (tileRows x tileCols) into LDS via the
// Tensor Data Mover. Descriptor per CDNA5 ISA 8.3/8.4 (D# group 0 + group 1).
// Wave-uniform; completion tracked by TENSORcnt (in-order per wave).
// ---------------------------------------------------------------------------
__device__ __forceinline__ void tdm_load_tile_f16(
    unsigned ldsAddr, const _Float16* gptr, unsigned tileCols,
    unsigned tileRows, unsigned rowStrideElems) {
  unsigned long long ga = (unsigned long long)(const void*)gptr;
  u32x4 g0;
  g0[0] = 1u;                                   // count=1, user descriptor
  g0[1] = ldsAddr;                              // lds_addr [63:32]
  g0[2] = (unsigned)(ga & 0xFFFFFFFFu);         // global_addr lo
  g0[3] = (unsigned)((ga >> 32) & 0x01FFFFFFu)  // global_addr hi (57-bit)
          | 0x80000000u;                        // type=2 ("image") [127:126]
  i32x8 g1;
  g1[0] = (int)(1u << 16);                      // wg_mask=0, data_size=1 (2B)
  g1[1] = (int)((rowStrideElems & 0xFFFFu) << 16);          // tensor_dim0 lo16
  g1[2] = (int)(((rowStrideElems >> 16) & 0xFFFFu)          // tensor_dim0 hi16
                | ((tileRows & 0xFFFFu) << 16));            // tensor_dim1 lo16
  g1[3] = (int)(((tileRows >> 16) & 0xFFFFu)                // tensor_dim1 hi16
                | ((tileCols & 0xFFFFu) << 16));            // tile_dim0
  g1[4] = (int)(tileRows & 0xFFFFu);                        // tile_dim1 (dim2=0)
  g1[5] = (int)rowStrideElems;                              // dim0_stride lo32
  g1[6] = 0;                                                // stride hi, dim1_stride lo
  g1[7] = 0;
  i32x4 z4 = {0, 0, 0, 0};                      // groups 2/3 unused (2D tensor)
  i32x8 z8 = {0, 0, 0, 0, 0, 0, 0, 0};
  __builtin_amdgcn_tensor_load_to_lds(g0, g1, z4, z4, z8, 0);
}

// ---------------------------------------------------------------------------
// fp32 -> f16 bulk convert (weights), 8 elements per thread
// ---------------------------------------------------------------------------
__global__ __launch_bounds__(256) void cvt_kernel(
    const float* __restrict__ src, _Float16* __restrict__ dst, long n) {
  long i = ((long)blockIdx.x * 256 + threadIdx.x) * 8;
  if (i < n) {
    float4 a = *(const float4*)(src + i);
    float4 b = *(const float4*)(src + i + 4);
    v8h o;
    o[0] = (_Float16)a.x; o[1] = (_Float16)a.y;
    o[2] = (_Float16)a.z; o[3] = (_Float16)a.w;
    o[4] = (_Float16)b.x; o[5] = (_Float16)b.y;
    o[6] = (_Float16)b.z; o[7] = (_Float16)b.w;
    *(v8h*)(dst + i) = o;
  }
}

// ---------------------------------------------------------------------------
// Embedding: x[t, d] = tok_emb[tokens[t], d] + pos_emb[t % L, d]
// ---------------------------------------------------------------------------
__global__ __launch_bounds__(256) void embed_kernel(
    const int* __restrict__ tokens, const float* __restrict__ tok_emb,
    const float* __restrict__ pos_emb, float* __restrict__ x) {
  int t = blockIdx.x;
  int l = t & (LSEQ - 1);
  int tok = tokens[t];
  int d = threadIdx.x * 4;
  float4 a = *(const float4*)(tok_emb + (size_t)tok * DIMN + d);
  float4 p = *(const float4*)(pos_emb + (size_t)l * DIMN + d);
  float4 r{a.x + p.x, a.y + p.y, a.z + p.z, a.w + p.w};
  *(float4*)(x + (size_t)t * DIMN + d) = r;
}

// ---------------------------------------------------------------------------
// LayerNorm over D=1024, f16 output. One block (256 threads) per token.
// ---------------------------------------------------------------------------
__global__ __launch_bounds__(256) void ln_kernel(
    const float* __restrict__ x, const float* __restrict__ g,
    const float* __restrict__ b, _Float16* __restrict__ out) {
  int t = blockIdx.x;
  const float* row = x + (size_t)t * DIMN;
  int d = threadIdx.x * 4;
  float4 v = *(const float4*)(row + d);
  float s  = v.x + v.y + v.z + v.w;
  float s2 = v.x * v.x + v.y * v.y + v.z * v.z + v.w * v.w;
  for (int m = 16; m >= 1; m >>= 1) {
    s  += __shfl_xor(s,  m, 32);
    s2 += __shfl_xor(s2, m, 32);
  }
  __shared__ float ws1[8], ws2[8];
  int w = threadIdx.x >> 5;
  if ((threadIdx.x & 31) == 0) { ws1[w] = s; ws2[w] = s2; }
  __syncthreads();
  float tot = 0.f, tot2 = 0.f;
  for (int i = 0; i < 8; ++i) { tot += ws1[i]; tot2 += ws2[i]; }
  float mean = tot * (1.0f / DIMN);
  float var  = tot2 * (1.0f / DIMN) - mean * mean;
  float inv  = rsqrtf(var + LN_EPS);
  float4 gg = *(const float4*)(g + d);
  float4 bb = *(const float4*)(b + d);
  _Float16* o = out + (size_t)t * DIMN + d;
  o[0] = (_Float16)((v.x - mean) * inv * gg.x + bb.x);
  o[1] = (_Float16)((v.y - mean) * inv * gg.y + bb.y);
  o[2] = (_Float16)((v.z - mean) * inv * gg.z + bb.z);
  o[3] = (_Float16)((v.w - mean) * inv * gg.w + bb.w);
}

// ---------------------------------------------------------------------------
// WMMA GEMM: Y[M x N] = A[M x K](f16) @ W16[N x K](f16)^T
// Block: 256 thr = 8 waves (4 m x 2 n), out tile 64x32, BK=64 (2 WMMA/wave/iter).
// Double-buffered TDM pipeline: wave 0 issues tensor_load_to_lds for tile k+1
// while all waves run WMMA on tile k; s_wait_tensorcnt(2) retires the oldest
// A/B descriptor pair (tensor ops complete in order per wave).
// OUT_MODE 0: store f16   OUT_MODE 1: f32 +=  (residual accumulate)
// ---------------------------------------------------------------------------
template <int OUT_MODE>
__global__ __launch_bounds__(256) void gemm_kernel(
    const _Float16* __restrict__ A, const _Float16* __restrict__ W16,
    void* __restrict__ Out, int M, int N, int K) {
  __shared__ _Float16 As[2][64][64];
  __shared__ _Float16 Bs[2][32][64];
  int tid  = threadIdx.x;
  int wave = tid >> 5, lane = tid & 31;
  int mw = wave & 3, nw = wave >> 2;
  int half = lane >> 4, ln16 = lane & 15;
  long mBase = (long)blockIdx.x * 64;
  long nBase = (long)blockIdx.y * 32;

  unsigned asAddr[2] = {(unsigned)(unsigned long long)(&As[0][0][0]),
                        (unsigned)(unsigned long long)(&As[1][0][0])};
  unsigned bsAddr[2] = {(unsigned)(unsigned long long)(&Bs[0][0][0]),
                        (unsigned)(unsigned long long)(&Bs[1][0][0])};
  const _Float16* aTile = A   + mBase * (long)K;
  const _Float16* bTile = W16 + nBase * (long)K;

  int m = mw * 16 + ln16;
  int n = nw * 16 + ln16;
  int nK = K >> 6;

  if (wave == 0) {                       // prologue: tile 0 in flight
    tdm_load_tile_f16(asAddr[0], aTile, 64, 64, (unsigned)K);
    tdm_load_tile_f16(bsAddr[0], bTile, 64, 32, (unsigned)K);
  }

  v8f c = {};
  for (int kt = 0; kt < nK; ++kt) {
    int cur = kt & 1;
    if (kt + 1 < nK) {
      if (wave == 0) {                   // issue next tile into other buffer
        tdm_load_tile_f16(asAddr[cur ^ 1], aTile + (kt + 1) * 64, 64, 64, (unsigned)K);
        tdm_load_tile_f16(bsAddr[cur ^ 1], bTile + (kt + 1) * 64, 64, 32, (unsigned)K);
      }
      __builtin_amdgcn_s_wait_tensorcnt(2);   // oldest A/B pair (tile kt) done
    } else {
      __builtin_amdgcn_s_wait_tensorcnt(0);   // last tile done
    }
    __syncthreads();

    for (int ch = 0; ch < 2; ++ch) {
      // A frag: lane = m%16 + 16*half; j<8 -> K=8*half+j, j>=8 -> 16+8*half+j-8
      v8h lo = *(const v8h*)&As[cur][m][ch * 32 + half * 8];
      v8h hi = *(const v8h*)&As[cur][m][ch * 32 + 16 + half * 8];
      v16h af;
      for (int j = 0; j < 8; ++j) { af[j] = lo[j]; af[j + 8] = hi[j]; }
      // B frag: N = ln16; j<8 -> K=8*half+j, j>=8 -> 16+8*half+j-8
      v8h blo = *(const v8h*)&Bs[cur][n][ch * 32 + half * 8];
      v8h bhi = *(const v8h*)&Bs[cur][n][ch * 32 + 16 + half * 8];
      v16h bf;
      for (int j = 0; j < 8; ++j) { bf[j] = blo[j]; bf[j + 8] = bhi[j]; }
      c = __builtin_amdgcn_wmma_f32_16x16x32_f16(false, af, false, bf,
                                                 (short)0, c, false, false);
    }
    __syncthreads();                      // buffer reusable for tile kt+2
  }

  long om = mBase + mw * 16 + half * 8;   // C layout: M = r + 8*half
  long on = nBase + nw * 16 + ln16;
  if (OUT_MODE == 0) {
    _Float16* o = (_Float16*)Out;
    for (int r = 0; r < 8; ++r) o[(om + r) * N + on] = (_Float16)c[r];
  } else {
    float* o = (float*)Out;
    for (int r = 0; r < 8; ++r) o[(om + r) * N + on] += c[r];
  }
}

// ---------------------------------------------------------------------------
// Per-head transpose of V: vT[(b*H+h)*64 + d][t] = v16[b*L+t][h*64+d]
// LDS-tiled so both global read and write are 32B-contiguous per lane.
// grid = (B*H, L/64), 256 threads.
// ---------------------------------------------------------------------------
__global__ __launch_bounds__(256) void vtr_kernel(
    const _Float16* __restrict__ v16, _Float16* __restrict__ vT) {
  __shared__ _Float16 tile[64][72];      // 72 halves = 144B row, 16B-aligned
  int bh = blockIdx.x;                   // b*HEADS + h
  int b  = bh >> 4, h = bh & 15;
  long t0 = (long)blockIdx.y * 64;
  int tid = threadIdx.x;
  int row = tid >> 2;                    // token within tile (0..63)
  int col = (tid & 3) * 16;              // head dim within tile
  *(v16h*)&tile[row][col] =
      *(const v16h*)(v16 + ((long)b * LSEQ + t0 + row) * DIMN + h * HEAD_DIM + col);
  __syncthreads();
  int d  = tid >> 2;                     // head dim (0..63)
  int tc = (tid & 3) * 16;               // token within tile
  v16h o;
  for (int j = 0; j < 16; ++j) o[j] = tile[tc + j][d];
  *(v16h*)(vT + ((long)bh * HEAD_DIM + d) * LSEQ + t0 + tc) = o;
}

// ---------------------------------------------------------------------------
// Flash attention, one wave per (b, h, 16-row q tile). 8 waves per block.
// q16/k16 are [T, DIMN]; vT is [B*H, HEAD_DIM, L] (head-transposed V).
// ---------------------------------------------------------------------------
__global__ __launch_bounds__(256) void attn_kernel(
    const _Float16* __restrict__ q16, const _Float16* __restrict__ k16,
    const _Float16* __restrict__ vT, _Float16* __restrict__ ao16) {
  __shared__ _Float16 pbuf[8][16][32];   // per-wave P transpose scratch
  int tid = threadIdx.x, wave = tid >> 5, lane = tid & 31;
  int half = lane >> 4, ln16 = lane & 15;
  int id  = blockIdx.x * 8 + wave;       // 0 .. B*H*(L/16)-1
  int b   = id >> 11;
  int rem = id & 2047;
  int h   = rem >> 7;
  int qt  = rem & 127;
  int qBase = qt * 16;
  long qRow0 = (long)b * LSEQ + qBase;
  int dOff = h * HEAD_DIM;
  const _Float16* vHead = vT + ((long)(b * HEADS + h) * HEAD_DIM) * LSEQ;

  // Q fragments for the two K32 chunks of HEAD_DIM
  v16h qa[2];
  for (int ch = 0; ch < 2; ++ch) {
    const _Float16* base = q16 + (qRow0 + ln16) * DIMN + dOff + ch * 32;
    v8h lo = *(const v8h*)(base + half * 8);
    v8h hi = *(const v8h*)(base + 16 + half * 8);
    v16h a;
    for (int j = 0; j < 8; ++j) { a[j] = lo[j]; a[j + 8] = hi[j]; }
    qa[ch] = a;
  }

  float rM[8], rL[8], o[4][8];
  for (int r = 0; r < 8; ++r) { rM[r] = -__builtin_inff(); rL[r] = 0.f; }
  for (int dt = 0; dt < 4; ++dt)
    for (int r = 0; r < 8; ++r) o[dt][r] = 0.f;

  int nKB = (qt >> 1) + 1;               // key blocks of 32 up to causal edge
  for (int jb = 0; jb < nKB; ++jb) {
    int kBase = jb * 32;
    if (jb + 1 < nKB) {                  // prefetch next key block (K and V^T)
      long nrow = (long)b * LSEQ + kBase + 32 + lane;
      __builtin_prefetch(k16 + nrow * DIMN + dOff, 0, 0);
      __builtin_prefetch(vHead + (long)lane * LSEQ + kBase + 32, 0, 0);
    }
    // S = Q K^T for two 16-col n tiles
    float sv[2][8];
    float bm[8];
    for (int r = 0; r < 8; ++r) bm[r] = -__builtin_inff();
    for (int nt = 0; nt < 2; ++nt) {
      v8f cc = {};
      long krow = (long)b * LSEQ + kBase + nt * 16 + ln16;
      for (int ch = 0; ch < 2; ++ch) {
        v8h blo = *(const v8h*)(k16 + krow * DIMN + dOff + ch * 32 + half * 8);
        v8h bhi = *(const v8h*)(k16 + krow * DIMN + dOff + ch * 32 + 16 + half * 8);
        v16h bf;
        for (int j = 0; j < 8; ++j) { bf[j] = blo[j]; bf[j + 8] = bhi[j]; }
        cc = __builtin_amdgcn_wmma_f32_16x16x32_f16(false, qa[ch], false, bf,
                                                    (short)0, cc, false, false);
      }
      int col = kBase + nt * 16 + ln16;
      for (int r = 0; r < 8; ++r) {
        int row = qBase + half * 8 + r;
        float s = cc[r] * ATT_SCALE;
        s = (col > row) ? -__builtin_inff() : (s + (float)col * 0.01f);
        sv[nt][r] = s;
        bm[r] = fmaxf(bm[r], s);
      }
    }
    for (int m = 1; m < 16; m <<= 1)
      for (int r = 0; r < 8; ++r) bm[r] = fmaxf(bm[r], __shfl_xor(bm[r], m, 32));

    float ps[8];
    for (int r = 0; r < 8; ++r) {
      float nm = fmaxf(rM[r], bm[r]);
      float corr = __expf(rM[r] - nm);
      rM[r] = nm;
      rL[r] *= corr;
      for (int dt = 0; dt < 4; ++dt) o[dt][r] *= corr;
      float acc = 0.f;
      for (int nt = 0; nt < 2; ++nt) {
        float p = __expf(sv[nt][r] - nm);
        sv[nt][r] = p;
        acc += p;
      }
      ps[r] = acc;
    }
    for (int m = 1; m < 16; m <<= 1)
      for (int r = 0; r < 8; ++r) ps[r] += __shfl_xor(ps[r], m, 32);
    for (int r = 0; r < 8; ++r) rL[r] += ps[r];

    // C-layout P -> LDS -> A-layout fragment (same wave, ds round trip)
    for (int nt = 0; nt < 2; ++nt) {
      int col = nt * 16 + ln16;
      for (int r = 0; r < 8; ++r)
        pbuf[wave][half * 8 + r][col] = (_Float16)sv[nt][r];
    }
    v16h pa;
    {
      v8h lo = *(const v8h*)&pbuf[wave][ln16][half * 8];
      v8h hi = *(const v8h*)&pbuf[wave][ln16][16 + half * 8];
      for (int j = 0; j < 8; ++j) { pa[j] = lo[j]; pa[j + 8] = hi[j]; }
    }

    // O += P @ V: four 16-col d tiles; B frag = two 16B loads of vT (K=8*half+j, 16+8*half+j)
    for (int dt = 0; dt < 4; ++dt) {
      const _Float16* vp = vHead + (long)(dt * 16 + ln16) * LSEQ + kBase;
      v8h vlo = *(const v8h*)(vp + half * 8);
      v8h vhi = *(const v8h*)(vp + 16 + half * 8);
      v16h vb;
      for (int j = 0; j < 8; ++j) { vb[j] = vlo[j]; vb[j + 8] = vhi[j]; }
      v8f cc;
      for (int r = 0; r < 8; ++r) cc[r] = o[dt][r];
      cc = __builtin_amdgcn_wmma_f32_16x16x32_f16(false, pa, false, vb,
                                                  (short)0, cc, false, false);
      for (int r = 0; r < 8; ++r) o[dt][r] = cc[r];
    }
  }

  for (int dt = 0; dt < 4; ++dt) {
    long col = dOff + dt * 16 + ln16;
    for (int r = 0; r < 8; ++r) {
      long row = qRow0 + half * 8 + r;
      ao16[row * DIMN + col] = (_Float16)(o[dt][r] / rL[r]);
    }
  }
}

// ---------------------------------------------------------------------------
// act = silu(f1) * fg, in place into f1 (8 halves per thread)
// ---------------------------------------------------------------------------
__global__ __launch_bounds__(256) void silu_gate_kernel(
    _Float16* __restrict__ f1, const _Float16* __restrict__ fg, long n) {
  long i = ((long)blockIdx.x * 256 + threadIdx.x) * 8;
  if (i < n) {
    v8h a = *(const v8h*)(f1 + i);
    v8h g = *(const v8h*)(fg + i);
    v8h o;
    for (int j = 0; j < 8; ++j) {
      float av = (float)a[j];
      float s  = av / (1.f + __expf(-av));
      o[j] = (_Float16)(s * (float)g[j]);
    }
    *(v8h*)(f1 + i) = o;
  }
}

// ---------------------------------------------------------------------------
// Final LN on last token + tiny [4 x 266] projection
// ---------------------------------------------------------------------------
__global__ __launch_bounds__(256) void final_kernel(
    const float* __restrict__ x, const float* __restrict__ lnf_g,
    const float* __restrict__ lnf_b, const float* __restrict__ lm_w,
    float* __restrict__ out) {
  int b = blockIdx.x;
  __shared__ float last[DIMN];
  const float* row = x + ((long)b * LSEQ + (LSEQ - 1)) * DIMN;
  int tid = threadIdx.x, d = tid * 4;
  float4 v = *(const float4*)(row + d);
  float s  = v.x + v.y + v.z + v.w;
  float s2 = v.x * v.x + v.y * v.y + v.z * v.z + v.w * v.w;
  for (int m = 16; m >= 1; m >>= 1) {
    s  += __shfl_xor(s,  m, 32);
    s2 += __shfl_xor(s2, m, 32);
  }
  __shared__ float ws1[8], ws2[8];
  if ((tid & 31) == 0) { ws1[tid >> 5] = s; ws2[tid >> 5] = s2; }
  __syncthreads();
  float tot = 0.f, tot2 = 0.f;
  for (int i = 0; i < 8; ++i) { tot += ws1[i]; tot2 += ws2[i]; }
  float mean = tot * (1.0f / DIMN);
  float var  = tot2 * (1.0f / DIMN) - mean * mean;
  float inv  = rsqrtf(var + LN_EPS);
  last[d + 0] = (v.x - mean) * inv * lnf_g[d + 0] + lnf_b[d + 0];
  last[d + 1] = (v.y - mean) * inv * lnf_g[d + 1] + lnf_b[d + 1];
  last[d + 2] = (v.z - mean) * inv * lnf_g[d + 2] + lnf_b[d + 2];
  last[d + 3] = (v.w - mean) * inv * lnf_g[d + 3] + lnf_b[d + 3];
  __syncthreads();
  for (int vv = tid; vv < VOCAB; vv += 256) {
    float acc = 0.f;
    const float* wrow = lm_w + (long)vv * DIMN;
    for (int k = 0; k < DIMN; ++k) acc += last[k] * wrow[k];
    out[b * VOCAB + vv] = acc;
  }
}

// ---------------------------------------------------------------------------
extern "C" void kernel_launch(void* const* d_in, const int* in_sizes, int n_in,
                              void* d_out, int out_size, void* d_ws, size_t ws_size,
                              hipStream_t stream) {
  const int*   tokens  = (const int*)  d_in[0];
  const float* tok_emb = (const float*)d_in[1];
  const float* pos_emb = (const float*)d_in[2];
  const float* ln1_g   = (const float*)d_in[3];
  const float* ln1_b   = (const float*)d_in[4];
  const float* q_w     = (const float*)d_in[5];
  const float* k_w     = (const float*)d_in[6];
  const float* v_w     = (const float*)d_in[7];
  const float* o_w     = (const float*)d_in[8];
  const float* ln2_g   = (const float*)d_in[9];
  const float* ln2_b   = (const float*)d_in[10];
  const float* w1_w    = (const float*)d_in[11];
  const float* wg_w    = (const float*)d_in[12];
  const float* w2_w    = (const float*)d_in[13];
  const float* lnf_g   = (const float*)d_in[14];
  const float* lnf_b   = (const float*)d_in[15];
  const float* lm_w    = (const float*)d_in[16];
  float* out = (float*)d_out;

  const long nProjW = (long)LAYERS * DIMN * DIMN;     // 8M
  const long nFfnW  = (long)LAYERS * HIDDEN * DIMN;   // 32M

  char* w = (char*)d_ws;
  float*     x    = (float*)w;      w += (size_t)TTOK * DIMN * sizeof(float);
  _Float16*  h16  = (_Float16*)w;   w += (size_t)TTOK * DIMN * 2;
  _Float16*  q16  = (_Float16*)w;   w += (size_t)TTOK * DIMN * 2;
  _Float16*  k16  = (_Float16*)w;   w += (size_t)TTOK * DIMN * 2;
  _Float16*  v16  = (_Float16*)w;   w += (size_t)TTOK * DIMN * 2;
  _Float16*  vT   = (_Float16*)w;   w += (size_t)TTOK * DIMN * 2;
  _Float16*  ao16 = (_Float16*)w;   w += (size_t)TTOK * DIMN * 2;
  _Float16*  f1   = (_Float16*)w;   w += (size_t)TTOK * HIDDEN * 2;
  _Float16*  fg   = (_Float16*)w;   w += (size_t)TTOK * HIDDEN * 2;
  _Float16*  qw16 = (_Float16*)w;   w += (size_t)nProjW * 2;
  _Float16*  kw16 = (_Float16*)w;   w += (size_t)nProjW * 2;
  _Float16*  vw16 = (_Float16*)w;   w += (size_t)nProjW * 2;
  _Float16*  ow16 = (_Float16*)w;   w += (size_t)nProjW * 2;
  _Float16*  w116 = (_Float16*)w;   w += (size_t)nFfnW * 2;
  _Float16*  wg16 = (_Float16*)w;   w += (size_t)nFfnW * 2;
  _Float16*  w216 = (_Float16*)w;   w += (size_t)nFfnW * 2;

  dim3 blk(256);
  // Weight conversion to f16 (TDM stages raw f16 tiles; no conversion in-flight)
  cvt_kernel<<<(int)(nProjW / 2048), blk, 0, stream>>>(q_w,  qw16, nProjW);
  cvt_kernel<<<(int)(nProjW / 2048), blk, 0, stream>>>(k_w,  kw16, nProjW);
  cvt_kernel<<<(int)(nProjW / 2048), blk, 0, stream>>>(v_w,  vw16, nProjW);
  cvt_kernel<<<(int)(nProjW / 2048), blk, 0, stream>>>(o_w,  ow16, nProjW);
  cvt_kernel<<<(int)(nFfnW  / 2048), blk, 0, stream>>>(w1_w, w116, nFfnW);
  cvt_kernel<<<(int)(nFfnW  / 2048), blk, 0, stream>>>(wg_w, wg16, nFfnW);
  cvt_kernel<<<(int)(nFfnW  / 2048), blk, 0, stream>>>(w2_w, w216, nFfnW);

  embed_kernel<<<TTOK, blk, 0, stream>>>(tokens, tok_emb, pos_emb, x);

  dim3 gProj(TTOK / 64, DIMN / 32);       // 128 x 32
  dim3 gFfn1(TTOK / 64, HIDDEN / 32);     // 128 x 128
  dim3 gVtr(BSZ * HEADS, LSEQ / 64);      // 64 x 32
  int  attnBlocks = BSZ * HEADS * (LSEQ / 16) / 8;   // 1024
  long nFfnElems = (long)TTOK * HIDDEN;

  for (int i = 0; i < LAYERS; ++i) {
    size_t po = (size_t)i * DIMN * DIMN;
    size_t fo = (size_t)i * HIDDEN * DIMN;
    ln_kernel<<<TTOK, blk, 0, stream>>>(x, ln1_g + (size_t)i * DIMN,
                                        ln1_b + (size_t)i * DIMN, h16);
    gemm_kernel<0><<<gProj, blk, 0, stream>>>(h16, qw16 + po, q16, TTOK, DIMN, DIMN);
    gemm_kernel<0><<<gProj, blk, 0, stream>>>(h16, kw16 + po, k16, TTOK, DIMN, DIMN);
    gemm_kernel<0><<<gProj, blk, 0, stream>>>(h16, vw16 + po, v16, TTOK, DIMN, DIMN);
    vtr_kernel<<<gVtr, blk, 0, stream>>>(v16, vT);
    attn_kernel<<<attnBlocks, blk, 0, stream>>>(q16, k16, vT, ao16);
    gemm_kernel<1><<<gProj, blk, 0, stream>>>(ao16, ow16 + po, x, TTOK, DIMN, DIMN);
    ln_kernel<<<TTOK, blk, 0, stream>>>(x, ln2_g + (size_t)i * DIMN,
                                        ln2_b + (size_t)i * DIMN, h16);
    gemm_kernel<0><<<gFfn1, blk, 0, stream>>>(h16, w116 + fo, f1, TTOK, HIDDEN, DIMN);
    gemm_kernel<0><<<gFfn1, blk, 0, stream>>>(h16, wg16 + fo, fg, TTOK, HIDDEN, DIMN);
    silu_gate_kernel<<<(int)(nFfnElems / 2048), blk, 0, stream>>>(f1, fg, nFfnElems);
    gemm_kernel<1><<<gProj, blk, 0, stream>>>(f1, w216 + fo, x, TTOK, DIMN, HIDDEN);
  }
  final_kernel<<<BSZ, blk, 0, stream>>>(x, lnf_g, lnf_b, lm_w, out);
}
